// ParaRNNBlock_83588653514756
// MI455X (gfx1250) — matrix operand order repaired and checked by reading, not compiled, not run on hardware
//
#include <hip/hip_runtime.h>

// Problem constants (from reference): B=4, S=4096, D=1024, K=4
#define BB 4
#define SS 4096
#define DD 1024
#define KK 4
#define EPSc 1e-6f
#define MROWS (BB * SS)        // 16384
#define NCHUNK 32
#define CHUNK 128              // SS / NCHUNK
#define LDSR 40                // LDS row stride in elems: 32 K + 8 pad (80 B, 16B-aligned)

typedef __attribute__((ext_vector_type(16))) __bf16 v16bf;
typedef __attribute__((ext_vector_type(8)))  __bf16 v8bf;
typedef __attribute__((ext_vector_type(8)))  float  v8f;
typedef __attribute__((ext_vector_type(8)))  unsigned short us8;
typedef __attribute__((ext_vector_type(4)))  unsigned int u32x4;
typedef __attribute__((ext_vector_type(8)))  int i32x8;
typedef __attribute__((ext_vector_type(4)))  int i32x4;

#define HAS_TDM __has_builtin(__builtin_amdgcn_tensor_load_to_lds)

__device__ __forceinline__ unsigned short f2bf(float f) {
    union { float f; unsigned int i; } v; v.f = f;
    unsigned int x = v.i;
    return (unsigned short)((x + 0x7FFFu + ((x >> 16) & 1u)) >> 16);  // RNE
}
__device__ __forceinline__ float sigm(float x) { return 1.f / (1.f + __expf(-x)); }
__device__ __forceinline__ unsigned ldsAddr(const void* p) {
    // generic LDS address: low 32 bits are the LDS byte offset (ISA 10.2 aperture rules)
    return (unsigned)(unsigned long long)p;
}

#if HAS_TDM
// Issue one TDM 2-D tile load: tile_d0 x tile_d1 elements (2B each) from a
// row-major tensor (row length tensor_d0, row stride stride0) into LDS at
// lds_byte_addr, padding +16B after every 64B row (LDS row stride = 80B).
// This toolchain's builtin is the 6-arg form:
//   (uint32x4 g0, int32x8 g1, int32x4 g2, int32x4 g3, int32x8 extra, i32 cpol)
__device__ __forceinline__ void tdm_load_2d(unsigned lds_byte_addr,
                                            const unsigned short* gptr,
                                            unsigned tensor_d0, unsigned tensor_d1,
                                            unsigned tile_d0, unsigned tile_d1,
                                            unsigned stride0) {
    unsigned long long ga = (unsigned long long)gptr;
    u32x4 g0;
    g0[0] = 1u;                                            // count=1 (valid), user desc
    g0[1] = lds_byte_addr;                                 // lds_addr
    g0[2] = (unsigned)ga;                                  // global_addr[31:0]
    g0[3] = (unsigned)((ga >> 32) & 0x01FFFFFFull)         // global_addr[56:32]
          | (2u << 30);                                    // type=2 ("image")
    i32x8 g1;
    g1[0] = (int)((1u << 16)                               // data_size = 2 bytes
                | (1u << 20)                               // pad_enable
                | (3u << 22)                               // pad_interval: 16 DWORDs (64B)
                | (3u << 25));                             // pad_amount: 4 DWORDs (16B)
    g1[1] = (int)((tensor_d0 & 0xFFFFu) << 16);            // tensor_dim0[15:0] -> bits 63:48
    g1[2] = (int)((tensor_d0 >> 16) | ((tensor_d1 & 0xFFFFu) << 16));
    g1[3] = (int)((tensor_d1 >> 16) | (tile_d0 << 16));    // tile_dim0 -> bits 127:112
    g1[4] = (int)(tile_d1);                                // tile_dim1 (tile_dim2=0)
    g1[5] = (int)stride0;                                  // tensor_dim0_stride[31:0]
    g1[6] = 0;                                             // stride0[47:32], stride1 lo
    g1[7] = 0;
    i32x4 z4 = {0, 0, 0, 0};
    i32x8 z8 = {0, 0, 0, 0, 0, 0, 0, 0};
    __builtin_amdgcn_tensor_load_to_lds(g0, g1, z4, z4, z8, 0);
}
#endif

// ---------------------------------------------------------------- weights fp32 -> bf16 (transposed)
// src: [Kdim, Ndim] row-major fp32  ->  dst: [Ndim, Kdim] row-major bf16
__global__ void f32_to_bf16_tr_kernel(const float* __restrict__ src,
                                      unsigned short* __restrict__ dst,
                                      int Kdim, int Ndim) {
    int i = blockIdx.x * blockDim.x + threadIdx.x;
    int n = i % Ndim, k = i / Ndim;
    if (k < Kdim) dst[(size_t)n * Kdim + k] = f2bf(src[(size_t)k * Ndim + n]);
}

// ---------------------------------------------------------------- RMSNorm(x)*w -> bf16
__global__ __launch_bounds__(256) void rmsnorm_bf16_kernel(
        const float* __restrict__ x, const float* __restrict__ w,
        unsigned short* __restrict__ out) {
    __shared__ float red[256];
    int row = blockIdx.x;
    const float* xr = x + (size_t)row * DD;
    float v[4], s = 0.f;
    #pragma unroll
    for (int i = 0; i < 4; i++) { v[i] = xr[threadIdx.x + 256 * i]; s += v[i] * v[i]; }
    red[threadIdx.x] = s; __syncthreads();
    for (int off = 128; off > 0; off >>= 1) {
        if ((int)threadIdx.x < off) red[threadIdx.x] += red[threadIdx.x + off];
        __syncthreads();
    }
    float rn = rsqrtf(red[0] * (1.f / DD) + EPSc);
    #pragma unroll
    for (int i = 0; i < 4; i++) {
        int d = threadIdx.x + 256 * i;
        out[(size_t)row * DD + d] = f2bf(v[i] * rn * w[d]);
    }
}

// ---------------------------------------------------------------- bf16 WMMA GEMM, TDM-staged
// C[M,N] = A[M,K](bf16, row-major) x Bt[N,K](bf16, row-major = B transposed), fp32 accum.
// Optional fused epilogue: C = resid * res_scale + acc.
// Block tile 128x128, 8 waves in 2x4 grid, each wave 64x32 = 4x2 WMMA 16x16 tiles.
// A and B tiles staged by the Tensor Data Mover, double-buffered, overlapped with WMMA.
__global__ __launch_bounds__(256) void gemm_bf16_wmma_kernel(
        const unsigned short* __restrict__ A, const unsigned short* __restrict__ Bt,
        float* __restrict__ C, const float* __restrict__ resid,
        const float* __restrict__ res_scale, int M, int N, int K) {
    __shared__ __align__(16) unsigned short As[2][128 * LDSR];
    __shared__ __align__(16) unsigned short Bs[2][128 * LDSR];

    const int tid  = threadIdx.x;
    const int lane = tid & 31, wid = tid >> 5;
    const int wm = wid & 1, wn = wid >> 1;        // 2 (M) x 4 (N) wave grid
    const int m0 = blockIdx.y * 128, n0 = blockIdx.x * 128;
    const int kh = lane >> 4, l16 = lane & 15;    // lane half / lane-in-half

    const v8f vzero = {0.f, 0.f, 0.f, 0.f, 0.f, 0.f, 0.f, 0.f};
    v8f acc[4][2];
    #pragma unroll
    for (int i = 0; i < 4; i++)
        #pragma unroll
        for (int j = 0; j < 2; j++) acc[i][j] = vzero;

    const unsigned short* Atile = A  + (size_t)m0 * K;   // rows m0.., cols k0..
    const unsigned short* Btile = Bt + (size_t)n0 * K;   // rows n0.., cols k0..
    const int nk = K / 32;

#if HAS_TDM
    if (wid == 0) {   // one wave per block drives the TDM
        tdm_load_2d(ldsAddr(&As[0][0]), Atile, K, M, 32, 128, K);
        tdm_load_2d(ldsAddr(&Bs[0][0]), Btile, K, N, 32, 128, K);
    }
#endif

    for (int it = 0; it < nk; it++) {
        const int buf = it & 1;
#if HAS_TDM
        if (wid == 0) {
            if (it + 1 < nk) {
                // prefetch next K-tile into the other buffer while this one computes
                tdm_load_2d(ldsAddr(&As[buf ^ 1][0]), Atile + (it + 1) * 32, K, M, 32, 128, K);
                tdm_load_2d(ldsAddr(&Bs[buf ^ 1][0]), Btile + (it + 1) * 32, K, N, 32, 128, K);
                __builtin_amdgcn_s_wait_tensorcnt(2);   // current tile's 2 DMAs done (in-order)
            } else {
                __builtin_amdgcn_s_wait_tensorcnt(0);
            }
        }
#else
        // Fallback: synchronous staging (both sources row-major, no transpose needed)
        {
            const int k0 = it * 32;
            #pragma unroll
            for (int i = 0; i < 2; i++) {
                int cc = tid + i * 256;                 // 512 chunks of 8 elems
                int row = cc >> 2, kc = (cc & 3) * 8;
                us8 da = *(const us8*)&Atile[(size_t)row * K + k0 + kc];
                *(us8*)&As[buf][row * LDSR + kc] = da;
                us8 db = *(const us8*)&Btile[(size_t)row * K + k0 + kc];
                *(us8*)&Bs[buf][row * LDSR + kc] = db;
            }
        }
#endif
        __syncthreads();   // LDS tile visible to all waves

        // ---- fragments, ISA 16-bit A layout:
        // lanes 0-15: K 0..7 & 16..23 ; lanes 16-31: K 8..15 & 24..31
        v16bf afrag[4], bfrag[2];
        #pragma unroll
        for (int tm = 0; tm < 4; tm++) {
            int r = wm * 64 + tm * 16 + l16;
            union { us8 u; v8bf b; } lo, hi;
            lo.u = *(const us8*)&As[buf][r * LDSR + kh * 8];
            hi.u = *(const us8*)&As[buf][r * LDSR + 16 + kh * 8];
            union { v16bf v; v8bf h[2]; } a; a.h[0] = lo.b; a.h[1] = hi.b;
            afrag[tm] = a.v;
        }
        #pragma unroll
        for (int tn = 0; tn < 2; tn++) {
            int nn = wn * 32 + tn * 16 + l16;
            union { us8 u; v8bf b; } lo, hi;
            lo.u = *(const us8*)&Bs[buf][nn * LDSR + kh * 8];
            hi.u = *(const us8*)&Bs[buf][nn * LDSR + 16 + kh * 8];
            union { v16bf v; v8bf h[2]; } b2; b2.h[0] = lo.b; b2.h[1] = hi.b;
            bfrag[tn] = b2.v;
        }

        #pragma unroll
        for (int tm = 0; tm < 4; tm++)
            #pragma unroll
            for (int tn = 0; tn < 2; tn++)
                acc[tm][tn] = __builtin_amdgcn_wmma_f32_16x16x32_bf16(
                    false, afrag[tm], false, bfrag[tn],
                    (short)0, acc[tm][tn], false, false);

        __syncthreads();   // all waves done with buf before TDM refills it next+1
    }

    // ---- epilogue: C/D layout -> VGPR r holds rows (r, r+8) split by lane half
    float rs = res_scale ? res_scale[0] : 0.f;
    #pragma unroll
    for (int tm = 0; tm < 4; tm++) {
        #pragma unroll
        for (int tn = 0; tn < 2; tn++) {
            int col = n0 + wn * 32 + tn * 16 + l16;
            int rowBase = m0 + wm * 64 + tm * 16 + kh * 8;
            #pragma unroll
            for (int r = 0; r < 8; r++) {
                size_t idx = (size_t)(rowBase + r) * N + col;
                float val = acc[tm][tn][r];
                if (resid) val = resid[idx] * rs + val;
                C[idx] = val;
            }
        }
    }
}

// ---------------------------------------------------------------- conv + silu + local scan
__global__ void conv_scan_phase1_kernel(
        const float* __restrict__ xz, const float* __restrict__ conv_w,
        const float* __restrict__ conv_b, const float* __restrict__ decay,
        float* __restrict__ hs, float* __restrict__ carries) {
    int tid = blockIdx.x * blockDim.x + threadIdx.x;   // B*D*NCHUNK threads
    int d = tid % DD;
    int c = (tid / DD) % NCHUNK;
    int b = tid / (DD * NCHUNK);
    float a  = sigm(decay[d]);
    float w0 = conv_w[d * KK + 0], w1 = conv_w[d * KK + 1];
    float w2 = conv_w[d * KK + 2], w3 = conv_w[d * KK + 3];
    float bias = conv_b[d];
    int t0 = c * CHUNK;
    const size_t strideT = 2 * DD;                     // xz row stride; xb = first half
    const float* xb = xz + (size_t)b * SS * strideT + d;
    float xm1 = (t0 >= 1) ? xb[(size_t)(t0 - 1) * strideT] : 0.f;
    float xm2 = (t0 >= 2) ? xb[(size_t)(t0 - 2) * strideT] : 0.f;
    float xm3 = (t0 >= 3) ? xb[(size_t)(t0 - 3) * strideT] : 0.f;
    float h = 0.f;
    float* hrow = hs + (size_t)b * SS * DD + d;
    for (int t = t0; t < t0 + CHUNK; t++) {
        float xt = xb[(size_t)t * strideT];
        float xc = fmaf(w3, xt, fmaf(w2, xm1, fmaf(w1, xm2, fmaf(w0, xm3, bias))));
        xm3 = xm2; xm2 = xm1; xm1 = xt;
        float v = xc * sigm(xc);                       // SiLU
        h = fmaf(a, h, v);
        hrow[(size_t)t * DD] = h;
    }
    carries[((size_t)b * NCHUNK + c) * DD + d] = h;
}

// ---------------------------------------------------------------- sequential carry combine
__global__ void scan_combine_kernel(float* __restrict__ carries,
                                    const float* __restrict__ decay) {
    int tid = blockIdx.x * blockDim.x + threadIdx.x;   // B*D threads
    int d = tid % DD, b = tid / DD;
    float a = sigm(decay[d]);
    float aL = a;
    #pragma unroll
    for (int i = 0; i < 7; i++) aL = aL * aL;          // a^128
    float cin = 0.f;
    for (int c = 0; c < NCHUNK; c++) {
        size_t idx = ((size_t)b * NCHUNK + c) * DD + d;
        float co = carries[idx];
        carries[idx] = cin;                            // incoming carry for chunk c
        cin = fmaf(aL, cin, co);
    }
}

// ---------------------------------------------------------------- apply carries
__global__ void scan_phase2_kernel(float* __restrict__ hs,
                                   const float* __restrict__ carries,
                                   const float* __restrict__ decay) {
    int tid = blockIdx.x * blockDim.x + threadIdx.x;   // B*D*NCHUNK threads
    int d = tid % DD;
    int c = (tid / DD) % NCHUNK;
    int b = tid / (DD * NCHUNK);
    float a = sigm(decay[d]);
    float cin = carries[((size_t)b * NCHUNK + c) * DD + d];
    float p = a;
    float* hrow = hs + (size_t)b * SS * DD + d;
    int t0 = c * CHUNK;
    for (int t = t0; t < t0 + CHUNK; t++) {
        hrow[(size_t)t * DD] = fmaf(p, cin, hrow[(size_t)t * DD]);
        p *= a;
    }
}

// ---------------------------------------------------------------- RMSNorm-gate * silu(z) -> bf16
__global__ __launch_bounds__(256) void gate_kernel(
        const float* __restrict__ hs, const float* __restrict__ xz,
        const float* __restrict__ gate_w, unsigned short* __restrict__ y) {
    __shared__ float red[256];
    int row = blockIdx.x;
    const float* hr = hs + (size_t)row * DD;
    float v[4], s = 0.f;
    #pragma unroll
    for (int i = 0; i < 4; i++) { v[i] = hr[threadIdx.x + 256 * i]; s += v[i] * v[i]; }
    red[threadIdx.x] = s; __syncthreads();
    for (int off = 128; off > 0; off >>= 1) {
        if ((int)threadIdx.x < off) red[threadIdx.x] += red[threadIdx.x + off];
        __syncthreads();
    }
    float rn = rsqrtf(red[0] * (1.f / DD) + EPSc);
    const float* zr = xz + (size_t)row * 2 * DD + DD;  // second half of xz
    #pragma unroll
    for (int i = 0; i < 4; i++) {
        int d = threadIdx.x + 256 * i;
        float z = zr[d];
        float g = v[i] * rn * gate_w[d] * (z * sigm(z));
        y[(size_t)row * DD + d] = f2bf(g);
    }
}

// ================================================================ launch
extern "C" void kernel_launch(void* const* d_in, const int* in_sizes, int n_in,
                              void* d_out, int out_size, void* d_ws, size_t ws_size,
                              hipStream_t stream) {
    (void)in_sizes; (void)n_in; (void)out_size; (void)ws_size;
    const float* x         = (const float*)d_in[0];
    const float* norm_w    = (const float*)d_in[1];
    const float* in_w      = (const float*)d_in[2];
    const float* conv_w    = (const float*)d_in[3];
    const float* conv_b    = (const float*)d_in[4];
    const float* decay     = (const float*)d_in[5];
    const float* gate_w    = (const float*)d_in[6];
    const float* out_w     = (const float*)d_in[7];
    const float* res_scale = (const float*)d_in[8];
    float* out = (float*)d_out;

    char* ws = (char*)d_ws;
    size_t off = 0;
    unsigned short* hbf      = (unsigned short*)(ws + off); off += (size_t)MROWS * DD * 2;      // 32 MB (reused for y)
    float*          xz       = (float*)(ws + off);          off += (size_t)MROWS * 2 * DD * 4;  // 128 MB
    float*          hs       = (float*)(ws + off);          off += (size_t)MROWS * DD * 4;      // 64 MB
    unsigned short* inw_bt   = (unsigned short*)(ws + off); off += (size_t)DD * 2 * DD * 2;     // 4 MB  [2D, D]
    unsigned short* outw_bt  = (unsigned short*)(ws + off); off += (size_t)DD * DD * 2;         // 2 MB  [D, D]
    float*          carries  = (float*)(ws + off);          off += (size_t)BB * NCHUNK * DD * 4;

    // 1) weights -> bf16, transposed to [N, K] so both GEMM operands are K-major
    f32_to_bf16_tr_kernel<<<(DD * 2 * DD) / 256, 256, 0, stream>>>(in_w, inw_bt, DD, 2 * DD);
    f32_to_bf16_tr_kernel<<<(DD * DD) / 256, 256, 0, stream>>>(out_w, outw_bt, DD, DD);
    // 2) RMSNorm -> bf16 activations
    rmsnorm_bf16_kernel<<<MROWS, 256, 0, stream>>>(x, norm_w, hbf);
    // 3) GEMM1: [16384,1024] x [1024,2048] -> xz fp32
    gemm_bf16_wmma_kernel<<<dim3((2 * DD) / 128, MROWS / 128), 256, 0, stream>>>(
        hbf, inw_bt, xz, nullptr, nullptr, MROWS, 2 * DD, DD);
    // 4-6) conv + silu + chunked linear scan
    conv_scan_phase1_kernel<<<(BB * DD * NCHUNK) / 256, 256, 0, stream>>>(
        xz, conv_w, conv_b, decay, hs, carries);
    scan_combine_kernel<<<(BB * DD) / 256, 256, 0, stream>>>(carries, decay);
    scan_phase2_kernel<<<(BB * DD * NCHUNK) / 256, 256, 0, stream>>>(hs, carries, decay);
    // 7) gate: rmsnorm(hs)*gate_w * silu(z) -> bf16 (reuse hbf)
    gate_kernel<<<MROWS, 256, 0, stream>>>(hs, xz, gate_w, hbf);
    // 8) GEMM2 with fused residual: out = x*res_scale + y @ out_w
    gemm_bf16_wmma_kernel<<<dim3(DD / 128, MROWS / 128), 256, 0, stream>>>(
        hbf, outw_bt, out, x, res_scale, MROWS, DD, DD);
}
